// ActionSeparatedLinearNN_50010599195074
// MI455X (gfx1250) — matrix operand order repaired and checked by reading, not compiled, not run on hardware
//
#include <hip/hip_runtime.h>
#include <hip/hip_bf16.h>

typedef __attribute__((ext_vector_type(16))) _Float16 v16h;
typedef __attribute__((ext_vector_type(8)))  _Float16 v8h;
typedef __attribute__((ext_vector_type(8)))  float    v8f;

#define NB 16384
#define NA 16
#define ND 512
#define NH 512
#define TM 128   // rows per workgroup (4 waves x 32 rows)
#define SP 520   // padded LDS row stride in f16 elements (breaks bank conflicts, keeps 16B align)

// ---------- grouping: histogram / scan / scatter ----------
__global__ void aslnn_init(int* counts, int* cursors) {
  int t = threadIdx.x;
  if (t < NA) { counts[t] = 0; cursors[t] = 0; }
}

__global__ void aslnn_hist(const int* __restrict__ act, int* __restrict__ counts) {
  int i = blockIdx.x * blockDim.x + threadIdx.x;
  if (i < NB) atomicAdd(&counts[act[i] & (NA - 1)], 1);
}

__global__ void aslnn_scan(const int* __restrict__ counts, int* __restrict__ offsets) {
  if (threadIdx.x == 0) {
    int s = 0;
    for (int i = 0; i < NA; ++i) { offsets[i] = s; s += counts[i]; }
  }
}

__global__ void aslnn_scatter(const int* __restrict__ act, const int* __restrict__ offsets,
                              int* __restrict__ cursors, int* __restrict__ perm) {
  int i = blockIdx.x * blockDim.x + threadIdx.x;
  if (i < NB) {
    int a = act[i] & (NA - 1);
    int p = atomicAdd(&cursors[a], 1);
    perm[offsets[a] + p] = i;
  }
}

// ---------- fp32 -> f16 conversion of state ----------
__global__ void aslnn_cvt(const float* __restrict__ in, _Float16* __restrict__ outp, int n4) {
  int i = blockIdx.x * blockDim.x + threadIdx.x;
  if (i < n4) {
    float4 v = ((const float4*)in)[i];
    union { _Float16 h[4]; unsigned long long u; } p;
    p.h[0] = (_Float16)v.x; p.h[1] = (_Float16)v.y;
    p.h[2] = (_Float16)v.z; p.h[3] = (_Float16)v.w;
    ((unsigned long long*)outp)[i] = p.u;
  }
}

// ---------- tiled transpose + convert: Wt[a][n][k] = (f16) W[a][k][n] ----------
__global__ void aslnn_tcvt(const float* __restrict__ in, _Float16* __restrict__ outp) {
  __shared__ float tile[32][33];
  int a  = blockIdx.z;
  int kb = blockIdx.x << 5;
  int nb = blockIdx.y << 5;
  const float* src = in + (size_t)a * ND * NH;
  _Float16* dst = outp + (size_t)a * ND * NH;
  int tx = threadIdx.x, ty = threadIdx.y;   // 32 x 8
  #pragma unroll
  for (int i = 0; i < 32; i += 8)
    tile[ty + i][tx] = src[(size_t)(kb + ty + i) * NH + nb + tx];
  __syncthreads();
  #pragma unroll
  for (int i = 0; i < 32; i += 8)
    dst[(size_t)(nb + ty + i) * ND + kb + tx] = (_Float16)tile[tx][ty + i];
}

// ---------- fused 3-layer MLP on one 128-row tile of one action group ----------
__global__ __launch_bounds__(128) void aslnn_mlp(
    const _Float16* __restrict__ stateH,
    const _Float16* __restrict__ W1t, const float* __restrict__ b1,
    const _Float16* __restrict__ W2t, const float* __restrict__ b2,
    const float* __restrict__ W3, const float* __restrict__ b3,
    const int* __restrict__ perm, const int* __restrict__ offsets,
    const int* __restrict__ counts, float* __restrict__ out)
{
  __shared__ __align__(16) _Float16 bufA[TM * SP];
  __shared__ __align__(16) _Float16 bufH[TM * SP];
  __shared__ int rowsIdx[TM];

  const int a = blockIdx.y;
  const int n_a = counts[a];
  const int tbase = blockIdx.x * TM;
  if (tbase >= n_a) return;
  const int off = offsets[a];
  int valid = n_a - tbase; if (valid > TM) valid = TM;

  const int t = threadIdx.x;
  const int lane = t & 31;
  const int wave = t >> 5;

  if (t < TM) {
    int rr = tbase + t;
    rowsIdx[t] = perm[off + (rr < n_a ? rr : tbase)];
  }
  __syncthreads();

  // stage state tile (f16) into LDS
  for (int i = t; i < TM * (ND / 8); i += 128) {
    int r = i >> 6;
    int c = (i & 63) << 3;
    uint4 v = *(const uint4*)(stateH + (size_t)rowsIdx[r] * ND + c);
    *(uint4*)(&bufA[r * SP + c]) = v;
  }
  __syncthreads();

  const int mrow = lane & 15;     // A row / B col / D col within 16-tile
  const int hi = lane >> 4;       // half-wave selector
  const int rowBase = wave * 32;  // this wave's 32 rows (two M fragments)

  auto layer = [&](const _Float16* src, _Float16* dst,
                   const _Float16* Wt, const float* bias) {
    for (int c = 0; c < NH / 64; ++c) {
      const int n0 = c * 64;
      float bv[4];
      #pragma unroll
      for (int j = 0; j < 4; ++j) bv[j] = bias[n0 + j * 16 + mrow];
      v8f acc[2][4];
      #pragma unroll
      for (int m = 0; m < 2; ++m)
        #pragma unroll
        for (int j = 0; j < 4; ++j) acc[m][j] = (v8f){};
      // Bounded unroll: keep a rolled, software-pipelined loop so the 8
      // independent accumulators issue interleaved (no dependent WMMA chains,
      // no VGPR-MSB spill into the 800-register range).
      #pragma unroll 2
      for (int k0 = 0; k0 < ND; k0 += 32) {
        // A fragments: lane holds row (rowBase + m*16 + mrow),
        // K = k0+hi*16 .. +15 (contiguous)
        union { v16h v; v8h h[2]; } af[2];
        #pragma unroll
        for (int m = 0; m < 2; ++m) {
          const _Float16* ap = src + (rowBase + m * 16 + mrow) * SP + k0 + hi * 16;
          af[m].h[0] = *(const v8h*)ap;
          af[m].h[1] = *(const v8h*)(ap + 8);
        }
        // B fragments: lane holds col (n0 + j*16 + mrow), K = k0+hi*16 .. +15 (contiguous)
        const _Float16* bp = Wt + (size_t)(n0 + mrow) * ND + k0 + hi * 16;
        v16h bf[4];
        #pragma unroll
        for (int j = 0; j < 4; ++j) bf[j] = *(const v16h*)(bp + j * 16 * ND);
        #pragma unroll
        for (int j = 0; j < 4; ++j) {
          acc[0][j] = __builtin_amdgcn_wmma_f32_16x16x32_f16(false, af[0].v, false, bf[j], (short)0, acc[0][j], false, false);
          acc[1][j] = __builtin_amdgcn_wmma_f32_16x16x32_f16(false, af[1].v, false, bf[j], (short)0, acc[1][j], false, false);
        }
      }
      // D layout: VGPR r -> row rowBase + m*16 + r + 8*hi, col n0 + j*16 + mrow
      #pragma unroll
      for (int m = 0; m < 2; ++m) {
        const int drow = rowBase + m * 16 + hi * 8;
        #pragma unroll
        for (int r = 0; r < 8; ++r) {
          _Float16* dp = dst + (drow + r) * SP + n0 + mrow;
          #pragma unroll
          for (int j = 0; j < 4; ++j) {
            float v = acc[m][j][r] + bv[j];
            v = v > 0.f ? v : 0.f;
            dp[j * 16] = (_Float16)v;
          }
        }
      }
    }
  };

  // layer 1: state -> h1
  layer(bufA, bufH, W1t + (size_t)a * ND * NH, b1 + a * NH);
  __syncthreads();
  // layer 2: h1 -> h2 (reuse state buffer)
  layer(bufH, bufA, W2t + (size_t)a * NH * NH, b2 + a * NH);
  __syncthreads();

  // final H -> 1 dot product + bias: one thread per row
  {
    const float* w3 = W3 + a * NH;
    float s = 0.f;
    const _Float16* hp = bufA + t * SP;
    #pragma unroll 8
    for (int k = 0; k < NH; ++k) s += (float)hp[k] * w3[k];
    if (t < valid) out[rowsIdx[t]] = s + b3[a];
  }
}

extern "C" void kernel_launch(void* const* d_in, const int* in_sizes, int n_in,
                              void* d_out, int out_size, void* d_ws, size_t ws_size,
                              hipStream_t stream) {
  const float* state = (const float*)d_in[0];
  const float* W1 = (const float*)d_in[1];
  const float* b1 = (const float*)d_in[2];
  const float* W2 = (const float*)d_in[3];
  const float* b2 = (const float*)d_in[4];
  const float* W3 = (const float*)d_in[5];
  const float* b3 = (const float*)d_in[6];
  const int* actions = (const int*)d_in[7];
  float* out = (float*)d_out;

  char* ws = (char*)d_ws;
  int* counts  = (int*)(ws + 0);
  int* offsets = (int*)(ws + 64);
  int* cursors = (int*)(ws + 128);
  int* perm    = (int*)(ws + 256);
  _Float16* stateH = (_Float16*)(ws + 256 + (size_t)NB * 4);
  _Float16* W1t = (_Float16*)(ws + 256 + (size_t)NB * 4 + (size_t)NB * ND * 2);
  _Float16* W2t = W1t + (size_t)NA * ND * NH;

  aslnn_init<<<1, 32, 0, stream>>>(counts, cursors);
  aslnn_hist<<<NB / 256, 256, 0, stream>>>(actions, counts);
  aslnn_scan<<<1, 32, 0, stream>>>(counts, offsets);
  aslnn_scatter<<<NB / 256, 256, 0, stream>>>(actions, offsets, cursors, perm);

  aslnn_cvt<<<(NB * ND / 4) / 256, 256, 0, stream>>>(state, stateH, NB * ND / 4);
  dim3 tgrid(ND / 32, NH / 32, NA), tblk(32, 8);
  aslnn_tcvt<<<tgrid, tblk, 0, stream>>>(W1, W1t);
  aslnn_tcvt<<<tgrid, tblk, 0, stream>>>(W2, W2t);

  dim3 mgrid(NB / TM, NA);
  aslnn_mlp<<<mgrid, 128, 0, stream>>>(stateH, W1t, b1, W2t, b2, W3, b3,
                                       perm, offsets, counts, out);
}
